// NLBlockND_7730941132981
// MI455X (gfx1250) — matrix operand order repaired and checked by reading, not compiled, not run on hardware
//
#include <hip/hip_runtime.h>
#include <hip/hip_bf16.h>

// ---------------------------------------------------------------------------
// NLBlockND (embedded-gaussian non-local attention) for MI455X / gfx1250.
// B=2, C=256, T=8,H=32,W=32 -> N=8192, INTER=64.
//   0) prep_weights : one-shot f32 -> bf16 conversion of all weight matrices.
//   1) proj_kernel  : x^T tile -> LDS (bf16, transposed) -> WMMA bf16 GEMMs
//                     producing theta/phi/g as (B,N,64) bf16 rows.
//   2) attn_kernel  : flash-attention (online softmax), S=Q*K^T and O=P*V via
//                     v_wmma_f32_16x16x32_bf16; 128-key blocks staged in
//                     dynamic LDS (66.8 KB, fine on 320KB/WGP CDNA5).
//   3) wz_kernel    : y*(Wz^T) via WMMA, writes wy (B,C,N) f32 + per-channel
//                     sum/sumsq (LDS reduce -> global f32 atomics).
//   4) bn_finalize + epilogue: batch-stats BN, gamma/beta, residual add.
// ---------------------------------------------------------------------------

typedef __bf16 bf16x16 __attribute__((ext_vector_type(16)));
typedef __bf16 bf16x2  __attribute__((ext_vector_type(2)));
typedef float  f32x8   __attribute__((ext_vector_type(8)));
typedef unsigned int   u32;
typedef unsigned short u16;

#define BDIM 2
#define CDIM 256
#define NDIM 8192
#define DDIM 64

struct FragBF { union { u32 u[8]; bf16x16 v; }; };

// Hardware conversion: fptrunc -> single v_cvt op.
__device__ __forceinline__ u16 f2bf(float x) {
  __bf16 h = (__bf16)x;
  return __builtin_bit_cast(u16, h);
}

__device__ __forceinline__ u32 pack_bf2(float a, float b) {
#if __has_builtin(__builtin_amdgcn_cvt_pk_bf16_f32)
  bf16x2 p = __builtin_amdgcn_cvt_pk_bf16_f32(a, b);
  return __builtin_bit_cast(u32, p);
#else
  return (u32)f2bf(a) | ((u32)f2bf(b) << 16);
#endif
}

__device__ __forceinline__ f32x8 zero8() {
  f32x8 z;
#pragma unroll
  for (int r = 0; r < 8; ++r) z[r] = 0.0f;
  return z;
}

// A-matrix 16x32 bf16 fragment from a row-major (K-contiguous) source.
// lane: row m = lane&15 (caller picks the row pointer), h = lane>>4.
// VGPR v<4 : K = kb + 8h + 2v ; VGPR v>=4 : K = kb + 16 + 8h + 2(v-4).
__device__ __forceinline__ FragBF load_frag_a(const u32* row32, int kb2, int h) {
  FragBF f;
#pragma unroll
  for (int v = 0; v < 4; ++v) {
    f.u[v]     = row32[kb2 + 4 * h + v];
    f.u[v + 4] = row32[kb2 + 8 + 4 * h + v];
  }
  return f;
}

// B-matrix 32x16 bf16 fragment; pointer is the row holding this lane's output
// column, K contiguous.  VGPR v<4 : K = kb + 8h + 2v ; v>=4 : K = kb + 16 + 8h + 2(v-4).
__device__ __forceinline__ FragBF load_frag_b(const u32* col32, int kb2, int h) {
  FragBF f;
#pragma unroll
  for (int v = 0; v < 4; ++v) {
    f.u[v]     = col32[kb2 + 4 * h + v];
    f.u[v + 4] = col32[kb2 + 8 + 4 * h + v];
  }
  return f;
}

__device__ __forceinline__ f32x8 wmma_bf16(const FragBF& a, const FragBF& b, f32x8 c) {
  return __builtin_amdgcn_wmma_f32_16x16x32_bf16(false, a.v, false, b.v,
                                                 (short)0, c, false, false);
}

// ---------------------------------------------------------------------------
// Pass 0: one-shot weight conversion to bf16.
// ---------------------------------------------------------------------------
__global__ __launch_bounds__(256) void prep_weights(
    const float* __restrict__ gw, const float* __restrict__ tw,
    const float* __restrict__ pw, const float* __restrict__ zw,
    u32* __restrict__ wg, u32* __restrict__ wt, u32* __restrict__ wp,
    u32* __restrict__ wz) {
  int i = blockIdx.x * 256 + threadIdx.x;  // 8192 u32 (16384 f32) per matrix
  float2 a = ((const float2*)gw)[i];
  wg[i] = pack_bf2(a.x, a.y);
  float2 b = ((const float2*)tw)[i];
  wt[i] = pack_bf2(b.x, b.y);
  float2 c = ((const float2*)pw)[i];
  wp[i] = pack_bf2(c.x, c.y);
  float2 d = ((const float2*)zw)[i];
  wz[i] = pack_bf2(d.x, d.y);
}

// ---------------------------------------------------------------------------
// Pass 1: theta/phi/g projections.  128 threads (4 waves), 64 positions/block.
// ---------------------------------------------------------------------------
#define XT_STRIDE 258   // 64 rows x 258 u16 (pad -> conflict-free transpose)

__global__ __launch_bounds__(128) void proj_kernel(
    const float* __restrict__ x,
    const u32* __restrict__ gw, const float* __restrict__ gb,
    const u32* __restrict__ tw, const float* __restrict__ tb,
    const u32* __restrict__ pw, const float* __restrict__ pb,
    u16* __restrict__ g_o, u16* __restrict__ th_o, u16* __restrict__ ph_o) {
  __shared__ __align__(16) u16 xT[64 * XT_STRIDE];

  const int b  = blockIdx.y;
  const int n0 = blockIdx.x * 64;
  const int t  = threadIdx.x;
  const float* xb = x + (size_t)b * CDIM * NDIM + n0;

#pragma unroll 8
  for (int i = 0; i < 32; ++i) {
    int idx = t + 128 * i;
    int c = idx >> 4, n4 = (idx & 15) * 4;      // coalesced along n
    float4 xv = *(const float4*)(xb + (size_t)c * NDIM + n4);
    xT[(n4 + 0) * XT_STRIDE + c] = f2bf(xv.x);
    xT[(n4 + 1) * XT_STRIDE + c] = f2bf(xv.y);
    xT[(n4 + 2) * XT_STRIDE + c] = f2bf(xv.z);
    xT[(n4 + 3) * XT_STRIDE + c] = f2bf(xv.w);
  }
  __syncthreads();

  const int wave = t >> 5, lane = t & 31;
  const int h = lane >> 4, nl = lane & 15;
  const u32* row32 = (const u32*)xT + (size_t)(16 * wave + nl) * (XT_STRIDE / 2);

  FragBF a[8];  // K = 256 = 8 x 32, reused across 3 proj x 4 nt
#pragma unroll
  for (int k = 0; k < 8; ++k) a[k] = load_frag_a(row32, 16 * k, h);

  const u32* W[3] = {gw, tw, pw};
  const float* Bv[3] = {gb, tb, pb};
  u16* O[3] = {g_o, th_o, ph_o};

#pragma unroll
  for (int j = 0; j < 3; ++j) {
    u16* ob = O[j] + (size_t)b * NDIM * DDIM;
#pragma unroll
    for (int nt = 0; nt < 4; ++nt) {
      const int o = nt * 16 + nl;
      const u32* bcol = W[j] + (size_t)o * (CDIM / 2);
      f32x8 acc = zero8();
#pragma unroll
      for (int k = 0; k < 8; ++k)
        acc = wmma_bf16(a[k], load_frag_b(bcol, 16 * k, h), acc);
      const float bias = Bv[j][o];
#pragma unroll
      for (int r = 0; r < 8; ++r) {
        int n = n0 + 16 * wave + r + 8 * h;
        ob[(size_t)n * DDIM + o] = f2bf(acc[r] + bias);
      }
    }
  }
}

// ---------------------------------------------------------------------------
// Pass 2: flash attention.  128 threads (4 waves), 128 queries/block,
// 128-key blocks.  Per wave: 32 query rows (2 M-tiles), 8 key N-tiles.
// Dynamic LDS: Kl 128x66 + Vt 64x130 + P 4x32x130 (u16) = 66,816 B.
// ---------------------------------------------------------------------------
#define BKEY 128
#define KL_STRIDE 66    // 33 u32, odd -> conflict-free
#define VT_STRIDE 130   // 65 u32, odd -> conflict-free

__global__ __launch_bounds__(128) void attn_kernel(
    const u16* __restrict__ theta, const u16* __restrict__ phi,
    const u16* __restrict__ g, u16* __restrict__ y) {
  extern __shared__ u16 smem[];
  u16* Kl = smem;                         // [key 0..127][d 0..63]   stride 66
  u16* Vt = Kl + BKEY * KL_STRIDE;        // [d 0..63][key 0..127]   stride 130
  u16* Pw = Vt + DDIM * VT_STRIDE;        // per-wave [32][128]      stride 130

  const int b  = blockIdx.y;
  const int n0 = blockIdx.x * 128;
  const int t  = threadIdx.x;
  const int wave = t >> 5, lane = t & 31;
  const int h = lane >> 4, nl = lane & 15;

  const u32* th32 = (const u32*)(theta + (size_t)b * NDIM * DDIM);
  const uint4* ph4 = (const uint4*)(phi + (size_t)b * NDIM * DDIM);
  const uint4* g4  = (const uint4*)(g   + (size_t)b * NDIM * DDIM);

  // Q fragments held in registers for the whole pass.
  FragBF qa[2][2];
#pragma unroll
  for (int mt = 0; mt < 2; ++mt)
#pragma unroll
    for (int ks = 0; ks < 2; ++ks)
      qa[mt][ks] = load_frag_a(th32 + (size_t)(n0 + 32 * wave + 16 * mt + nl) * 32,
                               16 * ks, h);

  f32x8 o[2][4];
  float rm[2][8], rs[2][8];
#pragma unroll
  for (int mt = 0; mt < 2; ++mt) {
#pragma unroll
    for (int nt = 0; nt < 4; ++nt) o[mt][nt] = zero8();
#pragma unroll
    for (int r = 0; r < 8; ++r) { rm[mt][r] = -3.0e38f; rs[mt][r] = 0.0f; }
  }

  u32* Kl32 = (u32*)Kl;
  u32* Vt32 = (u32*)Vt;
  u32* P32  = (u32*)(Pw + (size_t)wave * 32 * VT_STRIDE);
  u16* Pme  = Pw + (size_t)wave * 32 * VT_STRIDE;

  for (int kb = 0; kb < NDIM / BKEY; ++kb) {
    const int key0 = kb * BKEY;
    __syncthreads();
    // Stage K (keys x d, b128 loads) and V transposed [d][key].
#pragma unroll
    for (int i = 0; i < 8; ++i) {
      int idx = t + 128 * i;                     // 128 rows x 8 uint4
      int row = idx >> 3, q = idx & 7;
      uint4 kv = ph4[(size_t)(key0 + row) * 8 + q];
      u32* kd = Kl32 + row * (KL_STRIDE / 2) + q * 4;
      kd[0] = kv.x; kd[1] = kv.y; kd[2] = kv.z; kd[3] = kv.w;
      uint4 vv = g4[(size_t)(key0 + row) * 8 + q];
      int d0 = q * 8;
      Vt[(d0 + 0) * VT_STRIDE + row] = (u16)vv.x;
      Vt[(d0 + 1) * VT_STRIDE + row] = (u16)(vv.x >> 16);
      Vt[(d0 + 2) * VT_STRIDE + row] = (u16)vv.y;
      Vt[(d0 + 3) * VT_STRIDE + row] = (u16)(vv.y >> 16);
      Vt[(d0 + 4) * VT_STRIDE + row] = (u16)vv.z;
      Vt[(d0 + 5) * VT_STRIDE + row] = (u16)(vv.z >> 16);
      Vt[(d0 + 6) * VT_STRIDE + row] = (u16)vv.w;
      Vt[(d0 + 7) * VT_STRIDE + row] = (u16)(vv.w >> 16);
    }
    __syncthreads();

    if (kb + 1 < NDIM / BKEY) {  // prefetch next K/V block
      const char* pk = (const char*)(ph4 + (size_t)(key0 + BKEY) * 8);
      const char* pv = (const char*)(g4  + (size_t)(key0 + BKEY) * 8);
      __builtin_prefetch(pk + t * 128, 0, 1);
      __builtin_prefetch(pv + t * 128, 0, 1);
    }

    // S = Q * K^T  (2 M-tiles x 8 N-tiles x K=64); B-frags shared across mt.
    f32x8 s[2][8];
#pragma unroll
    for (int nt = 0; nt < 8; ++nt) {
      const u32* kcol = Kl32 + (size_t)(nt * 16 + nl) * (KL_STRIDE / 2);
      FragBF b0 = load_frag_b(kcol, 0, h);
      FragBF b1 = load_frag_b(kcol, 16, h);
#pragma unroll
      for (int mt = 0; mt < 2; ++mt) {
        f32x8 acc = wmma_bf16(qa[mt][0], b0, zero8());
        s[mt][nt] = wmma_bf16(qa[mt][1], b1, acc);
      }
    }

    // Online softmax per M-tile (rows r+8h, 16-lane shuffle reductions).
#pragma unroll
    for (int mt = 0; mt < 2; ++mt) {
      float tmax[8];
#pragma unroll
      for (int r = 0; r < 8; ++r) {
        float m = s[mt][0][r];
#pragma unroll
        for (int nt = 1; nt < 8; ++nt) m = fmaxf(m, s[mt][nt][r]);
#pragma unroll
        for (int msk = 1; msk < 16; msk <<= 1) m = fmaxf(m, __shfl_xor(m, msk));
        tmax[r] = m;
      }
#pragma unroll
      for (int r = 0; r < 8; ++r) {
        float nm = fmaxf(rm[mt][r], tmax[r]);
        float sc = __expf(rm[mt][r] - nm);
        rm[mt][r] = nm;
        rs[mt][r] *= sc;
#pragma unroll
        for (int nt = 0; nt < 4; ++nt) o[mt][nt][r] *= sc;
      }
      float psum[8];
#pragma unroll
      for (int r = 0; r < 8; ++r) psum[r] = 0.0f;
#pragma unroll
      for (int nt = 0; nt < 8; ++nt)
#pragma unroll
        for (int r = 0; r < 8; ++r) {
          float p = __expf(s[mt][nt][r] - rm[mt][r]);
          psum[r] += p;
          Pme[(16 * mt + r + 8 * h) * VT_STRIDE + nt * 16 + nl] = f2bf(p);
        }
#pragma unroll
      for (int r = 0; r < 8; ++r) {
        float q = psum[r];
#pragma unroll
        for (int msk = 1; msk < 16; msk <<= 1) q += __shfl_xor(q, msk);
        rs[mt][r] += q;
      }
    }

    // O += P * V  (K = 128 keys = 4 x 32); V-frags shared across mt.
#pragma unroll
    for (int ks = 0; ks < 4; ++ks) {
      FragBF pa0 = load_frag_a(P32 + (size_t)nl * (VT_STRIDE / 2), 16 * ks, h);
      FragBF pa1 = load_frag_a(P32 + (size_t)(16 + nl) * (VT_STRIDE / 2), 16 * ks, h);
#pragma unroll
      for (int dn = 0; dn < 4; ++dn) {
        const u32* vcol = Vt32 + (size_t)(dn * 16 + nl) * (VT_STRIDE / 2);
        FragBF vb = load_frag_b(vcol, 16 * ks, h);
        o[0][dn] = wmma_bf16(pa0, vb, o[0][dn]);
        o[1][dn] = wmma_bf16(pa1, vb, o[1][dn]);
      }
    }
  }

  // Normalize and write y (B,N,64) bf16.
  u16* yb = y + (size_t)b * NDIM * DDIM;
#pragma unroll
  for (int mt = 0; mt < 2; ++mt)
#pragma unroll
    for (int r = 0; r < 8; ++r) {
      float inv = 1.0f / rs[mt][r];
      int n = n0 + 32 * wave + 16 * mt + r + 8 * h;
#pragma unroll
      for (int nt = 0; nt < 4; ++nt)
        yb[(size_t)n * DDIM + nt * 16 + nl] = f2bf(o[mt][nt][r] * inv);
    }
}

// ---------------------------------------------------------------------------
// Pass 3a: wy = Wz * y^T + b (stored (B,C,N) f32) + per-channel sum/sumsq.
// ---------------------------------------------------------------------------
__global__ __launch_bounds__(256) void wz_kernel(
    const u16* __restrict__ y, const u32* __restrict__ wzw,
    const float* __restrict__ wzb, float* __restrict__ wy,
    float* __restrict__ gsum, float* __restrict__ gsq) {
  __shared__ float lsum[CDIM];
  __shared__ float lsq[CDIM];

  const int b  = blockIdx.y;
  const int n0 = blockIdx.x * 128;
  const int t  = threadIdx.x;
  const int wave = t >> 5, lane = t & 31;
  const int h = lane >> 4, nl = lane & 15;

  lsum[t] = 0.0f;
  lsq[t]  = 0.0f;
  __syncthreads();

  const u32* y32 = (const u32*)(y + (size_t)b * NDIM * DDIM);
  const u32* yrow = y32 + (size_t)(n0 + 16 * wave + nl) * 32;
  FragBF a0 = load_frag_a(yrow, 0, h);
  FragBF a1 = load_frag_a(yrow, 16, h);

#pragma unroll 4
  for (int nt = 0; nt < 16; ++nt) {
    const int c = nt * 16 + nl;
    const u32* bcol = wzw + (size_t)c * (DDIM / 2);
    f32x8 acc = zero8();
    acc = wmma_bf16(a0, load_frag_b(bcol, 0, h), acc);
    acc = wmma_bf16(a1, load_frag_b(bcol, 16, h), acc);
    const float bias = wzb[c];
    const int nb = n0 + 16 * wave + 8 * h;
    float* dst = wy + ((size_t)b * CDIM + c) * NDIM + nb;
    float4 lo, hi;
    lo.x = acc[0] + bias; lo.y = acc[1] + bias;
    lo.z = acc[2] + bias; lo.w = acc[3] + bias;
    hi.x = acc[4] + bias; hi.y = acc[5] + bias;
    hi.z = acc[6] + bias; hi.w = acc[7] + bias;
    float s8 = lo.x + lo.y + lo.z + lo.w + hi.x + hi.y + hi.z + hi.w;
    float q8 = lo.x * lo.x + lo.y * lo.y + lo.z * lo.z + lo.w * lo.w +
               hi.x * hi.x + hi.y * hi.y + hi.z * hi.z + hi.w * hi.w;
    *(float4*)(dst)     = lo;
    *(float4*)(dst + 4) = hi;
    atomicAdd(&lsum[c], s8);    // ds_add_f32
    atomicAdd(&lsq[c], q8);
  }
  __syncthreads();
  atomicAdd(&gsum[t], lsum[t]); // global_atomic_add_f32
  atomicAdd(&gsq[t], lsq[t]);
}

// ---------------------------------------------------------------------------
// Pass 3b/3c: finalize BN stats, apply scale/shift + residual.
// ---------------------------------------------------------------------------
__global__ void bn_finalize(const float* __restrict__ gsum,
                            const float* __restrict__ gsq,
                            const float* __restrict__ gamma,
                            const float* __restrict__ beta,
                            float* __restrict__ ab) {
  int c = threadIdx.x;
  const float cnt = (float)(BDIM * NDIM);
  float mean = gsum[c] / cnt;
  float var  = gsq[c] / cnt - mean * mean;
  float istd = rsqrtf(var + 1e-5f);
  float aa = gamma[c] * istd;
  ab[c]        = aa;
  ab[CDIM + c] = beta[c] - mean * aa;
}

__global__ __launch_bounds__(256) void epilogue(
    const float* __restrict__ wy, const float* __restrict__ ab,
    const float* __restrict__ x, float* __restrict__ out) {
  size_t i = ((size_t)blockIdx.x * 256 + threadIdx.x) * 4;
  int c = (int)((i >> 13) & (CDIM - 1));  // N=8192 floats per (b,c)
  float aa = ab[c], bb = ab[CDIM + c];
  float4 w  = *(const float4*)(wy + i);
  float4 xi = *(const float4*)(x + i);
  float4 o;
  o.x = fmaf(w.x, aa, bb) + xi.x;
  o.y = fmaf(w.y, aa, bb) + xi.y;
  o.z = fmaf(w.z, aa, bb) + xi.z;
  o.w = fmaf(w.w, aa, bb) + xi.w;
  *(float4*)(out + i) = o;
}

__global__ void zero512(float* p) { p[threadIdx.x] = 0.0f; }

// ---------------------------------------------------------------------------
extern "C" void kernel_launch(void* const* d_in, const int* in_sizes, int n_in,
                              void* d_out, int out_size, void* d_ws, size_t ws_size,
                              hipStream_t stream) {
  const float* x     = (const float*)d_in[0];
  const float* g_w   = (const float*)d_in[1];
  const float* g_b   = (const float*)d_in[2];
  const float* th_w  = (const float*)d_in[3];
  const float* th_b  = (const float*)d_in[4];
  const float* ph_w  = (const float*)d_in[5];
  const float* ph_b  = (const float*)d_in[6];
  const float* wz_w  = (const float*)d_in[7];
  const float* wz_b  = (const float*)d_in[8];
  const float* gamma = (const float*)d_in[9];
  const float* beta  = (const float*)d_in[10];
  float* out = (float*)d_out;

  const size_t BND = (size_t)BDIM * NDIM * DDIM;  // 1,048,576 elems
  u16* g_bf  = (u16*)d_ws;
  u16* th_bf = g_bf + BND;
  u16* ph_bf = th_bf + BND;
  u16* y_bf  = ph_bf + BND;
  float* wy   = (float*)((char*)d_ws + 4 * BND * sizeof(u16));   // 8 MB in
  float* gsum = (float*)((char*)wy + (size_t)BDIM * CDIM * NDIM * sizeof(float));
  float* gsq  = gsum + CDIM;
  float* ab   = gsq + CDIM;
  u32* wg_bf = (u32*)(ab + 2 * CDIM);   // bf16 weights, 8192 u32 each
  u32* wt_bf = wg_bf + 8192;
  u32* wp_bf = wt_bf + 8192;
  u32* wz_bf = wp_bf + 8192;

  zero512<<<1, 512, 0, stream>>>(gsum);  // gsum(256) + gsq(256)

  prep_weights<<<32, 256, 0, stream>>>(g_w, th_w, ph_w, wz_w,
                                       wg_bf, wt_bf, wp_bf, wz_bf);

  dim3 gp(NDIM / 64, BDIM);
  proj_kernel<<<gp, 128, 0, stream>>>(x, wg_bf, g_b, wt_bf, th_b, wp_bf, ph_b,
                                      g_bf, th_bf, ph_bf);

  // Dynamic LDS: Kl(128x66) + Vt(64x130) + P(4x32x130), u16.
  const size_t attn_lds =
      (size_t)(BKEY * KL_STRIDE + DDIM * VT_STRIDE + 4 * 32 * VT_STRIDE) *
      sizeof(u16);  // 66,816 bytes
  dim3 ga(NDIM / 128, BDIM);
  attn_kernel<<<ga, 128, attn_lds, stream>>>(th_bf, ph_bf, g_bf, y_bf);

  dim3 gz(NDIM / 128, BDIM);
  wz_kernel<<<gz, 256, 0, stream>>>(y_bf, wz_bf, wz_b, wy, gsum, gsq);

  bn_finalize<<<1, CDIM, 0, stream>>>(gsum, gsq, gamma, beta, ab);

  epilogue<<<(BDIM * CDIM * NDIM) / (256 * 4), 256, 0, stream>>>(wy, ab, x, out);
}
